// GATv2NodeGNN_48945447305700
// MI455X (gfx1250) — compile-verified, matrix operand order checked by reading, not run on hardware
//
#include <hip/hip_runtime.h>

#define IN_CH  128
#define D1     128   // HEADS*HID
#define HEADS  4
#define OUT_CH 64

typedef __attribute__((ext_vector_type(16))) _Float16 v16h;
typedef __attribute__((ext_vector_type(8)))  _Float16 v8h;
typedef __attribute__((ext_vector_type(8)))  float    v8f;

// ---- order-preserving float <-> uint for atomicMax-based segment max ----
__device__ __forceinline__ unsigned fenc(float f) {
  unsigned u = __float_as_uint(f);
  return (u & 0x80000000u) ? ~u : (u | 0x80000000u);
}
__device__ __forceinline__ float fdec(unsigned u) {
  return __uint_as_float((u & 0x80000000u) ? (u & 0x7FFFFFFFu) : ~u);
}

__global__ void fill_u32_kernel(unsigned* __restrict__ p, unsigned v, long n) {
  long i = (long)blockIdx.x * blockDim.x + threadIdx.x;
  if (i < n) p[i] = v;
}

__global__ void cvt_f16_kernel(const float* __restrict__ s, _Float16* __restrict__ d, long n) {
  long i = (long)blockIdx.x * blockDim.x + threadIdx.x;
  if (i < n) d[i] = (_Float16)s[i];
}

// W [K, Ncols] f32 row-major  ->  Wt [Ncols, K] f16 (columns become contiguous)
__global__ void transpose_cvt_kernel(const float* __restrict__ W, _Float16* __restrict__ Wt,
                                     int K, int Ncols) {
  int i = blockIdx.x * blockDim.x + threadIdx.x;
  if (i >= K * Ncols) return;
  int k = i / Ncols, n = i - k * Ncols;
  Wt[(size_t)n * K + k] = (_Float16)W[i];
}

// C[M,Ncols](f32) = A[M,K](f16, row-major) * Bt[Ncols,K](f16)^T, one wave per 16x16 tile.
__global__ void __launch_bounds__(32)
wmma_gemm_kernel(const _Float16* __restrict__ A, const _Float16* __restrict__ Bt,
                 float* __restrict__ C, int M, int K, int Ncols) {
  const int lane  = threadIdx.x & 31;
  const int row16 = blockIdx.x * 16;
  const int col16 = blockIdx.y * 16;
  const int r     = lane & 15;
  const int koff  = (lane < 16) ? 0 : 8;   // ISA 16-bit A/B fragment K sub-offset

  int ar = row16 + r; if (ar >= M) ar = M - 1;          // safe clamp for reads
  const _Float16* aRow = A  + (size_t)ar * K;
  const _Float16* bRow = Bt + (size_t)(col16 + r) * K;

  union HV { v16h v; v8h h[2]; };
  v8f acc = {};
  for (int k0 = 0; k0 < K; k0 += 32) {
    HV a, b;
    a.h[0] = *(const v8h*)(aRow + k0 + koff);
    a.h[1] = *(const v8h*)(aRow + k0 + 16 + koff);
    b.h[0] = *(const v8h*)(bRow + k0 + koff);
    b.h[1] = *(const v8h*)(bRow + k0 + 16 + koff);
    acc = __builtin_amdgcn_wmma_f32_16x16x32_f16(false, a.v, false, b.v,
                                                 (short)0, acc, false, false);
  }
  // D layout: VGPR j -> row (j | j+8), col = lane&15
  const int mbase = row16 + ((lane < 16) ? 0 : 8);
  float* cp = C + (size_t)mbase * Ncols + col16 + r;
#pragma unroll
  for (int j = 0; j < 8; ++j)
    if (mbase + j < M) cp[(size_t)j * Ncols] = acc[j];
}

// ================= layer 1 edge kernels (H=4, C=32, D1=128) =================
// one wave per edge; lane owns 4 contiguous channels (all in one head: head = lane>>3)
__global__ void edge_logits1_kernel(const float* __restrict__ xl, const float* __restrict__ xr,
                                    const int* __restrict__ srcA, const int* __restrict__ dstA,
                                    const float* __restrict__ att, float* __restrict__ logits,
                                    unsigned* __restrict__ nmax, int E, int ET) {
  long t = (long)blockIdx.x * blockDim.x + threadIdx.x;
  int e = (int)(t >> 5), lane = (int)(t & 31);
  if (e >= ET) return;
  int s, d;
  if (e < E) { s = srcA[e]; d = dstA[e]; } else { s = e - E; d = s; }
  const float4 a4 = *(const float4*)(xl + (size_t)s * D1 + lane * 4);
  const float4 b4 = *(const float4*)(xr + (size_t)d * D1 + lane * 4);
  const float4 w4 = *(const float4*)(att + lane * 4);
  float p = 0.f, v;
  v = a4.x + b4.x; v = v > 0.f ? v : 0.2f * v; p += v * w4.x;
  v = a4.y + b4.y; v = v > 0.f ? v : 0.2f * v; p += v * w4.y;
  v = a4.z + b4.z; v = v > 0.f ? v : 0.2f * v; p += v * w4.z;
  v = a4.w + b4.w; v = v > 0.f ? v : 0.2f * v; p += v * w4.w;
  p += __shfl_xor(p, 1, 32);
  p += __shfl_xor(p, 2, 32);
  p += __shfl_xor(p, 4, 32);
  if ((lane & 7) == 0) {
    int h = lane >> 3;
    logits[(size_t)e * HEADS + h] = p;
    atomicMax(nmax + (size_t)d * HEADS + h, fenc(p));
  }
}

__global__ void edge_exp1_kernel(const int* __restrict__ dstA, float* __restrict__ logits,
                                 const unsigned* __restrict__ nmax, float* __restrict__ nsum,
                                 int E, int ET) {
  long i = (long)blockIdx.x * blockDim.x + threadIdx.x;
  if (i >= (long)ET * HEADS) return;
  int e = (int)(i >> 2), h = (int)(i & 3);
  int d = (e < E) ? dstA[e] : e - E;
  float a = __expf(logits[i] - fdec(nmax[(size_t)d * HEADS + h]));
  logits[i] = a;
  atomicAdd(nsum + (size_t)d * HEADS + h, a);
}

__global__ void edge_agg1_kernel(const float* __restrict__ xl, const int* __restrict__ srcA,
                                 const int* __restrict__ dstA, const float* __restrict__ logits,
                                 const float* __restrict__ nsum, float* __restrict__ agg,
                                 int E, int ET) {
  long t = (long)blockIdx.x * blockDim.x + threadIdx.x;
  int e = (int)(t >> 5), lane = (int)(t & 31);
  if (e >= ET) return;
  int s, d;
  if (e < E) { s = srcA[e]; d = dstA[e]; } else { s = e - E; d = s; }
  int h = lane >> 3;
  float alpha = logits[(size_t)e * HEADS + h] / (nsum[(size_t)d * HEADS + h] + 1e-16f);
  const float4 a4 = *(const float4*)(xl + (size_t)s * D1 + lane * 4);
  float* o = agg + (size_t)d * D1 + lane * 4;
  atomicAdd(o + 0, a4.x * alpha);
  atomicAdd(o + 1, a4.y * alpha);
  atomicAdd(o + 2, a4.z * alpha);
  atomicAdd(o + 3, a4.w * alpha);
}

// h = elu(agg + b1) -> f16 (feeds layer-2 WMMA)
__global__ void elu_bias_cvt_kernel(const float* __restrict__ agg, const float* __restrict__ b,
                                    _Float16* __restrict__ hh, long n) {
  long i = (long)blockIdx.x * blockDim.x + threadIdx.x;
  if (i >= n) return;
  float v = agg[i] + b[i & (D1 - 1)];
  v = v > 0.f ? v : (__expf(v) - 1.f);
  hh[i] = (_Float16)v;
}

// ================= layer 2 edge kernels (H=1, C=64) =================
__global__ void edge_logits2_kernel(const float* __restrict__ xl, const float* __restrict__ xr,
                                    const int* __restrict__ srcA, const int* __restrict__ dstA,
                                    const float* __restrict__ att, float* __restrict__ logits,
                                    unsigned* __restrict__ nmax, int E, int ET) {
  long t = (long)blockIdx.x * blockDim.x + threadIdx.x;
  int e = (int)(t >> 5), lane = (int)(t & 31);
  if (e >= ET) return;
  int s, d;
  if (e < E) { s = srcA[e]; d = dstA[e]; } else { s = e - E; d = s; }
  const float2 a2 = *(const float2*)(xl + (size_t)s * OUT_CH + lane * 2);
  const float2 b2 = *(const float2*)(xr + (size_t)d * OUT_CH + lane * 2);
  const float2 w2 = *(const float2*)(att + lane * 2);
  float p = 0.f, v;
  v = a2.x + b2.x; v = v > 0.f ? v : 0.2f * v; p += v * w2.x;
  v = a2.y + b2.y; v = v > 0.f ? v : 0.2f * v; p += v * w2.y;
  p += __shfl_xor(p, 1, 32);
  p += __shfl_xor(p, 2, 32);
  p += __shfl_xor(p, 4, 32);
  p += __shfl_xor(p, 8, 32);
  p += __shfl_xor(p, 16, 32);
  if (lane == 0) {
    logits[e] = p;
    atomicMax(nmax + d, fenc(p));
  }
}

__global__ void edge_exp2_kernel(const int* __restrict__ dstA, float* __restrict__ logits,
                                 const unsigned* __restrict__ nmax, float* __restrict__ nsum,
                                 int E, int ET) {
  long i = (long)blockIdx.x * blockDim.x + threadIdx.x;
  if (i >= ET) return;
  int e = (int)i;
  int d = (e < E) ? dstA[e] : e - E;
  float a = __expf(logits[e] - fdec(nmax[d]));
  logits[e] = a;
  atomicAdd(nsum + d, a);
}

__global__ void edge_agg2_kernel(const float* __restrict__ xl, const int* __restrict__ srcA,
                                 const int* __restrict__ dstA, const float* __restrict__ logits,
                                 const float* __restrict__ nsum, float* __restrict__ out,
                                 int E, int ET) {
  long t = (long)blockIdx.x * blockDim.x + threadIdx.x;
  int e = (int)(t >> 5), lane = (int)(t & 31);
  if (e >= ET) return;
  int s, d;
  if (e < E) { s = srcA[e]; d = dstA[e]; } else { s = e - E; d = s; }
  float alpha = logits[e] / (nsum[d] + 1e-16f);
  const float2 a2 = *(const float2*)(xl + (size_t)s * OUT_CH + lane * 2);
  float* o = out + (size_t)d * OUT_CH + lane * 2;
  atomicAdd(o + 0, a2.x * alpha);
  atomicAdd(o + 1, a2.y * alpha);
}

__global__ void bias2_kernel(float* __restrict__ out, const float* __restrict__ b, long n) {
  long i = (long)blockIdx.x * blockDim.x + threadIdx.x;
  if (i < n) out[i] += b[i & (OUT_CH - 1)];
}

#define CDIV(a, b) (((a) + (b) - 1) / (b))

extern "C" void kernel_launch(void* const* d_in, const int* in_sizes, int n_in,
                              void* d_out, int out_size, void* d_ws, size_t ws_size,
                              hipStream_t stream) {
  const float* x    = (const float*)d_in[0];
  const int*   ei   = (const int*)d_in[1];
  const float* W1l  = (const float*)d_in[2];
  const float* W1r  = (const float*)d_in[3];
  const float* b1   = (const float*)d_in[4];
  const float* att1 = (const float*)d_in[5];
  const float* W2l  = (const float*)d_in[6];
  const float* W2r  = (const float*)d_in[7];
  const float* b2   = (const float*)d_in[8];
  const float* att2 = (const float*)d_in[9];

  const int N  = in_sizes[0] / IN_CH;   // 50000
  const int E  = in_sizes[1] / 2;       // 800000
  const int ET = E + N;                 // + self loops
  const int* srcA = ei;
  const int* dstA = ei + E;

  // ---- workspace carving (256B aligned) ----
  char* ws = (char*)d_ws;
  size_t o = 0;
  auto carve = [&](size_t bytes) { size_t r = o; o += (bytes + 255) & ~(size_t)255; return r; };
  _Float16* xh     = (_Float16*)(ws + carve((size_t)N * D1 * 2));        // x f16, later hh
  _Float16* W1lt   = (_Float16*)(ws + carve((size_t)IN_CH * D1 * 2));
  _Float16* W1rt   = (_Float16*)(ws + carve((size_t)IN_CH * D1 * 2));
  _Float16* W2lt   = (_Float16*)(ws + carve((size_t)D1 * OUT_CH * 2));
  _Float16* W2rt   = (_Float16*)(ws + carve((size_t)D1 * OUT_CH * 2));
  float*    xl1    = (float*)(ws + carve((size_t)N * D1 * 4));
  float*    xr1    = (float*)(ws + carve((size_t)N * D1 * 4));
  float*    logits = (float*)(ws + carve((size_t)ET * HEADS * 4));
  unsigned* nmax   = (unsigned*)(ws + carve((size_t)N * HEADS * 4));
  float*    nsum   = (float*)(ws + carve((size_t)N * HEADS * 4));
  float*    agg1   = (float*)(ws + carve((size_t)N * D1 * 4));
  // layer-2 aliases over dead layer-1 regions
  float* xl2 = xl1;                         // [N, 64]
  float* xr2 = xl1 + (size_t)N * OUT_CH;    // [N, 64]  (fits inside xl1's N*128)

  const int ew_blocks = (int)CDIV((long)ET * 32, 256);   // wave-per-edge launches

  // ---- prep: f16 conversions / transposes / accumulator zeroing ----
  long nx = (long)N * IN_CH;
  cvt_f16_kernel<<<(int)CDIV(nx, 256), 256, 0, stream>>>(x, xh, nx);
  transpose_cvt_kernel<<<CDIV(IN_CH * D1, 256), 256, 0, stream>>>(W1l, W1lt, IN_CH, D1);
  transpose_cvt_kernel<<<CDIV(IN_CH * D1, 256), 256, 0, stream>>>(W1r, W1rt, IN_CH, D1);
  transpose_cvt_kernel<<<CDIV(D1 * OUT_CH, 256), 256, 0, stream>>>(W2l, W2lt, D1, OUT_CH);
  transpose_cvt_kernel<<<CDIV(D1 * OUT_CH, 256), 256, 0, stream>>>(W2r, W2rt, D1, OUT_CH);
  fill_u32_kernel<<<(int)CDIV((long)N * HEADS, 256), 256, 0, stream>>>(nmax, 0u, (long)N * HEADS);
  fill_u32_kernel<<<(int)CDIV((long)N * HEADS, 256), 256, 0, stream>>>((unsigned*)nsum, 0u, (long)N * HEADS);
  fill_u32_kernel<<<(int)CDIV((long)N * D1, 256), 256, 0, stream>>>((unsigned*)agg1, 0u, (long)N * D1);

  // ---- layer 1 ----
  wmma_gemm_kernel<<<dim3(CDIV(N, 16), D1 / 16), 32, 0, stream>>>(xh, W1lt, xl1, N, IN_CH, D1);
  wmma_gemm_kernel<<<dim3(CDIV(N, 16), D1 / 16), 32, 0, stream>>>(xh, W1rt, xr1, N, IN_CH, D1);
  edge_logits1_kernel<<<ew_blocks, 256, 0, stream>>>(xl1, xr1, srcA, dstA, att1, logits, nmax, E, ET);
  edge_exp1_kernel<<<(int)CDIV((long)ET * HEADS, 256), 256, 0, stream>>>(dstA, logits, nmax, nsum, E, ET);
  edge_agg1_kernel<<<ew_blocks, 256, 0, stream>>>(xl1, srcA, dstA, logits, nsum, agg1, E, ET);
  elu_bias_cvt_kernel<<<(int)CDIV((long)N * D1, 256), 256, 0, stream>>>(agg1, b1, xh, (long)N * D1);

  // ---- layer 2 ----
  fill_u32_kernel<<<(int)CDIV((long)N, 256), 256, 0, stream>>>(nmax, 0u, (long)N);
  fill_u32_kernel<<<(int)CDIV((long)N, 256), 256, 0, stream>>>((unsigned*)nsum, 0u, (long)N);
  fill_u32_kernel<<<(int)CDIV((long)N * OUT_CH, 256), 256, 0, stream>>>((unsigned*)d_out, 0u, (long)N * OUT_CH);
  wmma_gemm_kernel<<<dim3(CDIV(N, 16), OUT_CH / 16), 32, 0, stream>>>(xh, W2lt, xl2, N, D1, OUT_CH);
  wmma_gemm_kernel<<<dim3(CDIV(N, 16), OUT_CH / 16), 32, 0, stream>>>(xh, W2rt, xr2, N, D1, OUT_CH);
  edge_logits2_kernel<<<ew_blocks, 256, 0, stream>>>(xl2, xr2, srcA, dstA, att2, logits, nmax, E, ET);
  edge_exp2_kernel<<<(int)CDIV((long)ET, 256), 256, 0, stream>>>(dstA, logits, nmax, nsum, E, ET);
  edge_agg2_kernel<<<ew_blocks, 256, 0, stream>>>(xl2, srcA, dstA, logits, nsum, (float*)d_out, E, ET);
  bias2_kernel<<<(int)CDIV((long)N * OUT_CH, 256), 256, 0, stream>>>((float*)d_out, b2, (long)N * OUT_CH);
}
